// DimeNet_41523743818101
// MI455X (gfx1250) — matrix operand order repaired and checked by reading, not compiled, not run on hardware
//
#include <hip/hip_runtime.h>
#include <math.h>

// ---------------------------------------------------------------------------
// DimeNet forward for gfx1250 (MI455X).  All EMB=128 contractions go through
// v_wmma_f32_16x16x32_bf16 (bf16 in, f32 accumulate).  Weight matrices are
// pre-packed once per launch into bf16 pairs laid out in this file's contiguous-k
// WMMA B-fragment order, then staged through LDS (ds_load_b128) in the GEMM.
// Geometry / narrow projections (NR=6, NS*NR=42) stay in scalar f32 VALU.
// Segment sums use global_atomic_add_f32.
// ---------------------------------------------------------------------------

#define EMB   128
#define NR    6
#define NS    7
#define NB    8
#define NBLK  2
#define NT    12
#define NG    64
#define CUTOFF_INV 0.2f

// packed weight: 8 col-tiles x 4 k-tiles x 32 lanes x 8 dwords = 8192 dwords
#define PKW   8192

typedef __attribute__((ext_vector_type(16))) __bf16        v16bf;
typedef __attribute__((ext_vector_type(8)))  float         v8f;
typedef __attribute__((ext_vector_type(8)))  unsigned int  v8u;

// GEMM epilogue flags
#define GF_SWISH  1   // y = swish(y)
#define GF_ADDOUT 2   // y = Y[idx] + y           (after activation)
#define GF_BIAS   4   // y += bias[col]           (before activation)
#define GF_PREADD 8   // y = acc + Y[idx]         (before bias/activation)

__constant__ float BROOTS[NS][NR] = {
    {3.141593f, 6.283185f, 9.424778f, 12.566371f, 15.707963f, 18.849556f},
    {4.493409f, 7.725252f, 10.904122f, 14.066194f, 17.220755f, 20.371303f},
    {5.763459f, 9.095011f, 12.322941f, 15.514603f, 18.689036f, 21.853874f},
    {6.987932f, 10.417119f, 13.698023f, 16.923621f, 20.121806f, 23.304246f},
    {8.182561f, 11.704907f, 15.039665f, 18.301256f, 21.525418f, 24.727566f},
    {9.355812f, 12.966530f, 16.354710f, 19.653152f, 22.904551f, 26.127750f},
    {10.512835f, 14.207392f, 17.647975f, 20.983463f, 24.262768f, 27.507868f}};

__constant__ float YNORM[NS] = {
    0.28209479f, 0.48860251f, 0.63078313f, 0.74635267f,
    0.84628438f, 0.93560257f, 1.01710723f};

__device__ __forceinline__ float swishf(float x) {
    return x / (1.0f + __expf(-x));
}

__device__ __forceinline__ float envelope_f(float x) {
    // p = 6 smooth cutoff: 1/x - 28 x^5 + 48 x^6 - 21 x^7   (0 for x >= 1)
    float x2 = x * x;
    float x5 = x2 * x2 * x;
    float e = 1.0f / x - 28.0f * x5 + 48.0f * x5 * x - 21.0f * x5 * x2;
    return (x < 1.0f) ? e : 0.0f;
}

__device__ float sph_jl(float x, int l) {
    float sx = sinf(x), cx = cosf(x);
    float inv = 1.0f / x;
    float jp = sx * inv;                   // j0
    if (l == 0) return jp;
    float jc = sx * inv * inv - cx * inv;  // j1
    for (int ll = 1; ll < l; ++ll) {
        float jn = (2.0f * ll + 1.0f) * inv * jc - jp;
        jp = jc;
        jc = jn;
    }
    return jc;
}

__device__ __forceinline__ unsigned int pack2bf(float lo, float hi) {
    unsigned short a = __builtin_bit_cast(unsigned short, (__bf16)lo);
    unsigned short b = __builtin_bit_cast(unsigned short, (__bf16)hi);
    return (unsigned int)a | ((unsigned int)b << 16);
}

// ---------------------------------------------------------------------------
// zero fill
// ---------------------------------------------------------------------------
__global__ void zero_kernel(float* __restrict__ p, size_t n) {
    size_t i = (size_t)blockIdx.x * blockDim.x + threadIdx.x;
    size_t stride = (size_t)gridDim.x * blockDim.x;
    for (; i < n; i += stride) p[i] = 0.0f;
}

// ---------------------------------------------------------------------------
// Pack a row-major [128,128] weight (k x col) into bf16 pairs laid out in
// contiguous-k WMMA B-fragment order:  P[((ct*4+kt)*32+lane)*8+v] = pack(W[k,col],W[k+1,col])
// with k = kt*32 + (lane/16)*16 + 2v, col = ct*16 + lane%16.   8192 threads.
// ---------------------------------------------------------------------------
__global__ void pack_w_kernel(const float* __restrict__ W, unsigned int* __restrict__ P) {
    int idx = blockIdx.x * blockDim.x + threadIdx.x;       // 0..8191
    int v = idx & 7, lane = (idx >> 3) & 31, kt = (idx >> 8) & 3, ct = idx >> 10;
    int k = kt * 32 + (lane >> 4) * 16 + 2 * v;
    int col = ct * 16 + (lane & 15);
    P[idx] = pack2bf(W[(size_t)k * EMB + col], W[(size_t)(k + 1) * EMB + col]);
}

// Pack Wbil [2,128(i),8(j),128(l)] into per-(b,j) B-fragments with the role
// B[k=l][col=i]:  P[b][(((j*8+ct)*4+kt)*32+lane)*8+v]
__global__ void pack_wbil_kernel(const float* __restrict__ Wbil, unsigned int* __restrict__ P) {
    int idx = blockIdx.x * blockDim.x + threadIdx.x;       // 0..(2*8*8192-1)
    int v = idx & 7, lane = (idx >> 3) & 31, kt = (idx >> 8) & 3;
    int ct = (idx >> 10) & 7, j = (idx >> 13) & 7, b = idx >> 16;
    int k = kt * 32 + (lane >> 4) * 16 + 2 * v;
    int i = ct * 16 + (lane & 15);
    const float* Wb = Wbil + (((size_t)b * EMB + i) * NB + j) * EMB;
    P[idx] = pack2bf(Wb[k], Wb[k + 1]);
}

// ---------------------------------------------------------------------------
// per-edge geometry: x = |R_i - R_j|/cutoff, envelope, rbf[E,6]
// ---------------------------------------------------------------------------
__global__ void edge_kernel(const float* __restrict__ R,
                            const int* __restrict__ src, const int* __restrict__ dst,
                            const float* __restrict__ freqs,
                            float* __restrict__ xe, float* __restrict__ rbf, int E) {
    int e = blockIdx.x * blockDim.x + threadIdx.x;
    if (e >= E) return;
    int s = src[e], d = dst[e];
    float dx = R[3 * d + 0] - R[3 * s + 0];
    float dy = R[3 * d + 1] - R[3 * s + 1];
    float dz = R[3 * d + 2] - R[3 * s + 2];
    float dist = sqrtf(dx * dx + dy * dy + dz * dz);
    float x = dist * CUTOFF_INV;
    xe[e] = x;
    float env = envelope_f(x);
#pragma unroll
    for (int r = 0; r < NR; ++r)
        rbf[(size_t)e * NR + r] = env * sinf(freqs[r] * x);
}

// rbf_e = swish(rbf @ W_rbf_e + b)   — grid E, block 128
__global__ void rbfe_kernel(const float* __restrict__ rbf,
                            const float* __restrict__ W, const float* __restrict__ B,
                            float* __restrict__ Y) {
    int e = blockIdx.x, c = threadIdx.x;
    float a = B[c];
#pragma unroll
    for (int r = 0; r < NR; ++r) a += rbf[(size_t)e * NR + r] * W[r * EMB + c];
    Y[(size_t)e * EMB + c] = swishf(a);
}

// h[n,:] = atom_emb[Z[n],:]
__global__ void gather_h_kernel(const float* __restrict__ atom_emb,
                                const int* __restrict__ Z, float* __restrict__ h) {
    int n = blockIdx.x, c = threadIdx.x;
    h[(size_t)n * EMB + c] = atom_emb[(size_t)Z[n] * EMB + c];
}

// xkj[e,c] *= rbf[e,:] @ Wi_rbf[:,c]
__global__ void mulrbf_kernel(float* __restrict__ xkj, const float* __restrict__ rbf,
                              const float* __restrict__ Wirbf) {
    int e = blockIdx.x, c = threadIdx.x;
    float p = 0.0f;
#pragma unroll
    for (int r = 0; r < NR; ++r) p += rbf[(size_t)e * NR + r] * Wirbf[r * EMB + c];
    xkj[(size_t)e * EMB + c] *= p;
}

// ---------------------------------------------------------------------------
// triplet geometry: cbf (Legendre * ynorm), sbf (spherical Bessel * env),
// projected through Wi_sbf for both interaction blocks -> sbfw[T,16]
// ---------------------------------------------------------------------------
__global__ void triplet_kernel(const float* __restrict__ R, const float* __restrict__ xe,
                               const int* __restrict__ src, const int* __restrict__ dst,
                               const int* __restrict__ t_kj, const int* __restrict__ t_ji,
                               const float* __restrict__ Wi_sbf,  // [2,42,8]
                               float* __restrict__ sbfw, int T) {
    int t = blockIdx.x * blockDim.x + threadIdx.x;
    if (t >= T) return;
    int eji = t_ji[t], ekj = t_kj[t];
    int j = src[eji], i = dst[eji], k = src[ekj];
    float v1x = R[3 * i + 0] - R[3 * j + 0];
    float v1y = R[3 * i + 1] - R[3 * j + 1];
    float v1z = R[3 * i + 2] - R[3 * j + 2];
    float v2x = R[3 * k + 0] - R[3 * j + 0];
    float v2y = R[3 * k + 1] - R[3 * j + 1];
    float v2z = R[3 * k + 2] - R[3 * j + 2];
    float n1 = sqrtf(v1x * v1x + v1y * v1y + v1z * v1z);
    float n2 = sqrtf(v2x * v2x + v2y * v2y + v2z * v2z);
    float ct = (v1x * v2x + v1y * v2y + v1z * v2z) / (n1 * n2);
    ct = fminf(1.0f, fmaxf(-1.0f, ct));

    float cb[NS];
    cb[0] = 1.0f;
    cb[1] = ct;
#pragma unroll
    for (int l = 1; l < NS - 1; ++l)
        cb[l + 1] = ((2.0f * l + 1.0f) * ct * cb[l] - (float)l * cb[l - 1]) / (float)(l + 1);
#pragma unroll
    for (int l = 0; l < NS; ++l) cb[l] *= YNORM[l];

    float xk = xe[ekj];
    float envk = envelope_f(xk);

    float sb[NS * NR];
    for (int l = 0; l < NS; ++l)
        for (int r = 0; r < NR; ++r)
            sb[l * NR + r] = sph_jl(xk * BROOTS[l][r], l) * envk * cb[l];

    for (int b = 0; b < NBLK; ++b) {
#pragma unroll
        for (int n = 0; n < NB; ++n) {
            float acc = 0.0f;
            for (int s = 0; s < NS * NR; ++s)
                acc += sb[s] * Wi_sbf[((size_t)b * (NS * NR) + s) * NB + n];
            sbfw[(size_t)t * 16 + b * NB + n] = acc;
        }
    }
}

// ---------------------------------------------------------------------------
// Generic [M,128] x [128,128] WMMA GEMM, bf16 inputs / f32 accumulate.
// Block = 256 threads = 8 waves; wave w owns col tile w; K=128 as 4x
// v_wmma_f32_16x16x32_bf16.  The pre-packed bf16 weight (32 KB) is staged
// in LDS cooperatively; each wave pulls its B fragment as one aligned
// 32-byte LDS read.  Optional row-gather on X via gidx.
// Fragment layouts as built here (contiguous k for A and B, not the ISA table order):
//   A(16x32 bf16): lane L -> row L%16; VGPR v -> K = (L/16)*16 + v*2
//   B(32x16 bf16): lane L -> col L%16; VGPR v -> K = (L/16)*16 + v*2
//   C/D(16x16 f32): lane L -> col L%16; VGPR v -> row v + 8*(L/16)
// ---------------------------------------------------------------------------
__global__ __launch_bounds__(256)
void gemm128(const float* __restrict__ X, const int* __restrict__ gidx,
             const unsigned int* __restrict__ Wp, const float* __restrict__ B,
             float* __restrict__ Y, int M, int flags) {
    __shared__ __align__(32) unsigned int lds_w[PKW];
    {
        const uint4* s4 = (const uint4*)Wp;
        uint4* d4 = (uint4*)lds_w;
#pragma unroll
        for (int i = 0; i < 8; ++i)
            d4[threadIdx.x + 256 * i] = s4[threadIdx.x + 256 * i];
    }
    __syncthreads();

    int wave = threadIdx.x >> 5;
    int lane = threadIdx.x & 31;
    int halfl = lane >> 4;
    int rl = lane & 15;
    int row0 = blockIdx.x * 16;
    int col0 = wave * 16;

    int arow = row0 + rl;
    if (arow >= M) arow = M - 1;           // clamp: keeps EXEC uniform for WMMA
    int xrow = gidx ? gidx[arow] : arow;
    const float* xr = X + (size_t)xrow * EMB;
    int colB = col0 + rl;

    v8f acc = {0.f, 0.f, 0.f, 0.f, 0.f, 0.f, 0.f, 0.f};
#pragma unroll
    for (int kt = 0; kt < 4; ++kt) {
        int kb = kt * 32;
        v16bf a;
#pragma unroll
        for (int v = 0; v < 8; ++v) {
            int ka = kb + halfl * 16 + v * 2;
            a[2 * v]     = (__bf16)xr[ka];
            a[2 * v + 1] = (__bf16)xr[ka + 1];
        }
        const v8u* bp = (const v8u*)&lds_w[((wave * 4 + kt) * 32 + lane) * 8];
        v16bf bf = __builtin_bit_cast(v16bf, *bp);
        acc = __builtin_amdgcn_wmma_f32_16x16x32_bf16(false, a, false, bf,
                                                      (short)0, acc, false, false);
    }

    float bv = (flags & GF_BIAS) ? B[colB] : 0.0f;
#pragma unroll
    for (int v = 0; v < 8; ++v) {
        int row = row0 + v + 8 * halfl;
        if (row < M) {
            size_t idx = (size_t)row * EMB + colB;
            float y = acc[v];
            if (flags & GF_PREADD) y += Y[idx];
            y += bv;
            if (flags & GF_SWISH) y = swishf(y);
            if (flags & GF_ADDOUT) y += Y[idx];
            Y[idx] = y;
        }
    }
}

// ---------------------------------------------------------------------------
// Bilinear triplet interaction:
//   msg[t,i] = sum_{j<8} sbf_w[t,j] * sum_l xkj[t_kj[t],l] * Wbil[i,j,l]
// A operands (f32) are loaded once and kept in VGPRs; per-j only the scalar
// scale + bf16 cvt is redone.  Wbil comes pre-packed in fragment order
// (contiguous b128 loads).  Result scatter-added into m2 (= x_ji) via t_ji.
// ---------------------------------------------------------------------------
__global__ __launch_bounds__(256)
void bilinear16(const float* __restrict__ xkj, const float* __restrict__ sbfw,
                const int* __restrict__ t_kj, const int* __restrict__ t_ji,
                const unsigned int* __restrict__ Pbil,  // [8 j][8 ct][4 kt][32][8]
                float* __restrict__ m2, int T, int sb_off) {
    int wave = threadIdx.x >> 5;
    int lane = threadIdx.x & 31;
    int halfl = lane >> 4;
    int rl = lane & 15;
    int row0 = blockIdx.x * 16;
    int col0 = wave * 16;

    int trow = row0 + rl;
    bool avalid = trow < T;
    int tcl = avalid ? trow : (T - 1);
    int ge = t_kj[tcl];
    const float* xr = xkj + (size_t)ge * EMB;
    int colB = col0 + rl;

    // preload the 64 j-invariant A operands (f32) into registers
    float ax[64];
#pragma unroll
    for (int kt = 0; kt < 4; ++kt) {
#pragma unroll
        for (int v = 0; v < 8; ++v) {
            int ka = kt * 32 + halfl * 16 + v * 2;
            ax[kt * 16 + 2 * v]     = xr[ka];
            ax[kt * 16 + 2 * v + 1] = xr[ka + 1];
        }
    }

    v8f acc = {0.f, 0.f, 0.f, 0.f, 0.f, 0.f, 0.f, 0.f};
#pragma unroll
    for (int j = 0; j < NB; ++j) {
        float s = avalid ? sbfw[(size_t)tcl * 16 + sb_off + j] : 0.0f;
#pragma unroll
        for (int kt = 0; kt < 4; ++kt) {
            v16bf a;
#pragma unroll
            for (int v = 0; v < 8; ++v) {
                a[2 * v]     = (__bf16)(s * ax[kt * 16 + 2 * v]);
                a[2 * v + 1] = (__bf16)(s * ax[kt * 16 + 2 * v + 1]);
            }
            const v8u* bp = (const v8u*)&Pbil[((size_t)((j * 8 + wave) * 4 + kt) * 32 + lane) * 8];
            v16bf bf = __builtin_bit_cast(v16bf, *bp);
            acc = __builtin_amdgcn_wmma_f32_16x16x32_bf16(false, a, false, bf,
                                                          (short)0, acc, false, false);
        }
    }
#pragma unroll
    for (int v = 0; v < 8; ++v) {
        int row = row0 + v + 8 * halfl;
        if (row < T) {
            int e = t_ji[row];
            atomicAdd(&m2[(size_t)e * EMB + colB], acc[v]);
        }
    }
}

// ---------------------------------------------------------------------------
// out-block stage 1: nodes[dst[e],c] += (rbf[e,:] @ Wo_rbf[:,c]) * m[e,c]
// ---------------------------------------------------------------------------
__global__ void out_stage1(const float* __restrict__ rbf, const float* __restrict__ Worbf,
                           const float* __restrict__ m, const int* __restrict__ dst,
                           float* __restrict__ nodes) {
    int e = blockIdx.x, c = threadIdx.x;
    float p = 0.0f;
#pragma unroll
    for (int r = 0; r < NR; ++r) p += rbf[(size_t)e * NR + r] * Worbf[r * EMB + c];
    atomicAdd(&nodes[(size_t)dst[e] * EMB + c], p * m[(size_t)e * EMB + c]);
}

// out-block final: out[batch_seg[n], c] += nodes[n,:] @ Wo_f[:,c]
__global__ void out_stage2(const float* __restrict__ nodes, const float* __restrict__ Wof,
                           const int* __restrict__ bseg, float* __restrict__ out) {
    int n = blockIdx.x, c = threadIdx.x;
    if (c >= NT) return;
    float a = 0.0f;
    for (int k = 0; k < EMB; ++k) a += nodes[(size_t)n * EMB + k] * Wof[k * NT + c];
    atomicAdd(&out[(size_t)bseg[n] * NT + c], a);
}

// ---------------------------------------------------------------------------
extern "C" void kernel_launch(void* const* d_in, const int* in_sizes, int n_in,
                              void* d_out, int out_size, void* d_ws, size_t ws_size,
                              hipStream_t stream) {
    (void)n_in; (void)out_size; (void)ws_size;
    const float* R        = (const float*)d_in[0];
    const float* freqs    = (const float*)d_in[1];
    const float* atom_emb = (const float*)d_in[2];
    const float* W_rbf_e  = (const float*)d_in[3];
    const float* b_rbf_e  = (const float*)d_in[4];
    const float* W_e      = (const float*)d_in[5];
    const float* b_e      = (const float*)d_in[6];
    const float* Wo_rbf   = (const float*)d_in[7];   // [3,6,128]
    const float* Wo_d     = (const float*)d_in[8];   // [3,3,128,128]
    const float* bo_d     = (const float*)d_in[9];   // [3,3,128]
    const float* Wo_f     = (const float*)d_in[10];  // [3,128,12]
    const float* Wi_rbf   = (const float*)d_in[11];  // [2,6,128]
    const float* Wi_sbf   = (const float*)d_in[12];  // [2,42,8]
    const float* Wji      = (const float*)d_in[13];
    const float* bji      = (const float*)d_in[14];
    const float* Wkj      = (const float*)d_in[15];
    const float* bkj      = (const float*)d_in[16];
    const float* Wbil     = (const float*)d_in[17];  // [2,128,8,128]
    const float* Wbs      = (const float*)d_in[18];  // [2,2,128,128]
    const float* bbs      = (const float*)d_in[19];
    const float* Wskip    = (const float*)d_in[20];
    const float* bskip    = (const float*)d_in[21];
    const float* Was      = (const float*)d_in[22];  // [2,4,128,128]
    const float* bas      = (const float*)d_in[23];
    const int* Z     = (const int*)d_in[24];
    const int* src   = (const int*)d_in[25];
    const int* dst   = (const int*)d_in[26];
    const int* t_kj  = (const int*)d_in[27];
    const int* t_ji  = (const int*)d_in[28];
    const int* bseg  = (const int*)d_in[29];

    const int N = in_sizes[24];
    const int E = in_sizes[25];
    const int T = in_sizes[27];

    float* out = (float*)d_out;

    // ---- workspace carve-out (f32 activations) ----
    float* wsf = (float*)d_ws;
    size_t off = 0;
    auto carve = [&](size_t n) { float* p = wsf + off; off += n; return p; };
    float* xe    = carve((size_t)E);
    float* rbf   = carve((size_t)E * NR);
    float* h     = carve((size_t)N * EMB);
    float* rbfe  = carve((size_t)E * EMB);
    float* m     = carve((size_t)E * EMB);
    float* m2    = carve((size_t)E * EMB);
    float* t1    = carve((size_t)E * EMB);
    float* xkj   = carve((size_t)E * EMB);
    float* sbfw  = carve((size_t)T * 16);
    float* nodes = carve((size_t)N * EMB);
    float* node2 = carve((size_t)N * EMB);

    // ---- packed bf16 weights (fragment order) ----
    unsigned int* pk = (unsigned int*)(wsf + off);
    auto packW = [&](const float* W) -> const unsigned int* {
        unsigned int* p = pk;
        pk += PKW;
        pack_w_kernel<<<PKW / 256, 256, 0, stream>>>(W, p);
        return p;
    };
    const size_t sq = (size_t)EMB * EMB;
    const unsigned int* pWe0 = packW(W_e);
    const unsigned int* pWe1 = packW(W_e + 128 * EMB);
    const unsigned int* pWe2 = packW(W_e + 256 * EMB);
    const unsigned int* pWod[3][3];
    for (int b = 0; b < 3; ++b)
        for (int kd = 0; kd < 3; ++kd)
            pWod[b][kd] = packW(Wo_d + ((size_t)b * 3 + kd) * sq);
    const unsigned int* pWji[NBLK];
    const unsigned int* pWkj[NBLK];
    const unsigned int* pWskip[NBLK];
    const unsigned int* pWbs[NBLK][2];
    const unsigned int* pWas[NBLK][4];
    for (int b = 0; b < NBLK; ++b) {
        pWji[b]   = packW(Wji + b * sq);
        pWkj[b]   = packW(Wkj + b * sq);
        pWskip[b] = packW(Wskip + b * sq);
        for (int r = 0; r < 2; ++r) pWbs[b][r] = packW(Wbs + ((size_t)b * 2 + r) * sq);
        for (int r = 0; r < 4; ++r) pWas[b][r] = packW(Was + ((size_t)b * 4 + r) * sq);
    }
    unsigned int* pWbil = pk;
    pk += (size_t)NBLK * NB * PKW;
    pack_wbil_kernel<<<(NBLK * NB * PKW) / 256, 256, 0, stream>>>(Wbil, pWbil);

    auto gemm = [&](const float* X, const int* gi, const unsigned int* Wp, const float* B,
                    float* Y, int M, int flags) {
        gemm128<<<dim3((M + 15) / 16), 256, 0, stream>>>(X, gi, Wp, B, Y, M, flags);
    };
    auto out_block = [&](int b, const float* min) {
        zero_kernel<<<2048, 256, 0, stream>>>(nodes, (size_t)N * EMB);
        out_stage1<<<E, EMB, 0, stream>>>(rbf, Wo_rbf + (size_t)b * NR * EMB, min, dst, nodes);
        gemm(nodes, nullptr, pWod[b][0], bo_d + ((size_t)b * 3 + 0) * EMB, node2, N,
             GF_SWISH | GF_BIAS);
        gemm(node2, nullptr, pWod[b][1], bo_d + ((size_t)b * 3 + 1) * EMB, nodes, N,
             GF_SWISH | GF_BIAS);
        gemm(nodes, nullptr, pWod[b][2], bo_d + ((size_t)b * 3 + 2) * EMB, node2, N,
             GF_SWISH | GF_BIAS);
        out_stage2<<<N, 16, 0, stream>>>(node2, Wo_f + (size_t)b * EMB * NT, bseg, out);
    };

    // --- geometry ---
    zero_kernel<<<4, 256, 0, stream>>>(out, (size_t)NG * NT);
    edge_kernel<<<(E + 255) / 256, 256, 0, stream>>>(R, src, dst, freqs, xe, rbf, E);
    rbfe_kernel<<<E, EMB, 0, stream>>>(rbf, W_rbf_e, b_rbf_e, rbfe);
    gather_h_kernel<<<N, EMB, 0, stream>>>(atom_emb, Z, h);
    triplet_kernel<<<(T + 127) / 128, 128, 0, stream>>>(R, xe, src, dst, t_kj, t_ji,
                                                        Wi_sbf, sbfw, T);

    // --- embedding: m = swish([h[src]|h[dst]|rbf_e] @ W_e + b_e) as 3 K=128 passes
    gemm(h,    src,     pWe0, nullptr, m, E, 0);
    gemm(h,    dst,     pWe1, nullptr, m, E, GF_PREADD);
    gemm(rbfe, nullptr, pWe2, b_e,     m, E, GF_PREADD | GF_BIAS | GF_SWISH);

    out_block(0, m);

    // --- interaction blocks ---
    for (int b = 0; b < NBLK; ++b) {
        // x_ji -> m2 ; x_kj -> xkj (then scaled by rbf @ Wi_rbf)
        gemm(m, nullptr, pWji[b], bji + (size_t)b * EMB, m2, E, GF_SWISH | GF_BIAS);
        gemm(m, nullptr, pWkj[b], bkj + (size_t)b * EMB, xkj, E, GF_SWISH | GF_BIAS);
        mulrbf_kernel<<<E, EMB, 0, stream>>>(xkj, rbf, Wi_rbf + (size_t)b * NR * EMB);
        // m2 += bilinear triplet aggregation (scatter via t_ji)
        bilinear16<<<(T + 15) / 16, 256, 0, stream>>>(xkj, sbfw, t_kj, t_ji,
                                                      pWbil + (size_t)b * NB * PKW,
                                                      m2, T, b * NB);
        // residual MLP on m2
        gemm(m2, nullptr, pWbs[b][0], bbs + ((size_t)b * 2 + 0) * EMB, t1, E,
             GF_SWISH | GF_BIAS);
        gemm(t1, nullptr, pWbs[b][1], bbs + ((size_t)b * 2 + 1) * EMB, m2, E,
             GF_SWISH | GF_BIAS | GF_ADDOUT);
        // m = swish(m2 @ Wskip + b) + m
        gemm(m2, nullptr, pWskip[b], bskip + (size_t)b * EMB, m, E,
             GF_SWISH | GF_BIAS | GF_ADDOUT);
        // two post-skip residual MLPs
        for (int r = 0; r < 2; ++r) {
            gemm(m, nullptr, pWas[b][2 * r], bas + ((size_t)b * 4 + 2 * r) * EMB, t1, E,
                 GF_SWISH | GF_BIAS);
            gemm(t1, nullptr, pWas[b][2 * r + 1], bas + ((size_t)b * 4 + 2 * r + 1) * EMB, m, E,
                 GF_SWISH | GF_BIAS | GF_ADDOUT);
        }
        out_block(b + 1, m);
    }
}
